// SingleModalAtten_6021544149579
// MI455X (gfx1250) — matrix operand order repaired and checked by reading, not compiled, not run on hardware
//
#include <hip/hip_runtime.h>

// ---------------------------------------------------------------------------
// SingleModalAtten for MI455X (gfx1250, wave32, WMMA bf16 -> fp32 accum)
// B=4, C=1024, T=2048, H=16 (head dim 64), GroupNorm groups G=32.
//
// v3: LDS strides padded to 72 elems (144B, 16B-aligned rows) + explicit
//     16B vector fragment loads -> ds_load_b128; software-pipelined global
//     loads (reg-staged) overlapping WMMA compute in GEMMs and attention.
//
// Workspace: [0,4KB) stats | [4KB,+16MB) xn bf16 | +48MB qkv bf16 | +16MB attn
// ---------------------------------------------------------------------------

typedef __bf16 bf16;
typedef __attribute__((ext_vector_type(16))) __bf16 v16bf;
typedef __attribute__((ext_vector_type(8)))  __bf16 v8bf;
typedef __attribute__((ext_vector_type(8)))  float  v8f;

#define NB 4
#define NC 1024
#define NT 2048
#define NH 16
#define CH 64
#define NG 32

__device__ inline v8f wmma_bf16(v16bf a, v16bf b, v8f c) {
    return __builtin_amdgcn_wmma_f32_16x16x32_bf16(
        false, a, false, b, (short)0, c, false, false);
}

__device__ inline v16bf combine16(v8bf lo, v8bf hi) {
    return __builtin_shufflevector(lo, hi, 0, 1, 2, 3, 4, 5, 6, 7,
                                   8, 9, 10, 11, 12, 13, 14, 15);
}

// A fragment: 16x32 (MxK), row-major LDS, stride LDA (LDA*2 must be 16B mult).
// lane half selects K base 0/8; element runs at +0..7 and +16..23.
template <int LDA>
__device__ inline v16bf load_a_frag(const bf16* As, int m0, int k0) {
    const int lane = threadIdx.x & 31;
    const int kb   = (lane < 16) ? 0 : 8;
    const bf16* row = As + (m0 + (lane & 15)) * LDA + k0 + kb;
    const v8bf lo = *(const v8bf*)(row);
    const v8bf hi = *(const v8bf*)(row + 16);
    return combine16(lo, hi);
}

// B fragment from N-major LDS tile Bt[N][K]: lane n, lane half selects K base
// 0/8; element runs at +0..7 and +16..23 -> exactly 2x ds_load_b128.
template <int LDB>
__device__ inline v16bf load_b_frag_t(const bf16* Bt, int k0, int n0) {
    const int lane = threadIdx.x & 31;
    const int kb   = (lane < 16) ? 0 : 8;
    const bf16* row = Bt + (n0 + (lane & 15)) * LDB + k0 + kb;
    const v8bf lo = *(const v8bf*)(row);
    const v8bf hi = *(const v8bf*)(row + 16);
    return combine16(lo, hi);
}

// ---------------------------------------------------------------------------
// Kernel 1: GroupNorm statistics (one block per (b,g); contiguous 32*T seg).
// ---------------------------------------------------------------------------
__global__ __launch_bounds__(256) void gn_stats_kernel(
    const float* __restrict__ x, float* __restrict__ stats) {
    const int seg = blockIdx.x;
    const float* p = x + (size_t)seg * (32u * NT);
    float s = 0.f, ss = 0.f;
    for (int i = threadIdx.x; i < 32 * NT; i += 256) {
        const float v = p[i];
        s += v;
        ss += v * v;
    }
    __shared__ float r0[256], r1[256];
    r0[threadIdx.x] = s;
    r1[threadIdx.x] = ss;
    __syncthreads();
    for (int off = 128; off > 0; off >>= 1) {
        if (threadIdx.x < off) {
            r0[threadIdx.x] += r0[threadIdx.x + off];
            r1[threadIdx.x] += r1[threadIdx.x + off];
        }
        __syncthreads();
    }
    if (threadIdx.x == 0) {
        const float mean = r0[0] * (1.0f / 65536.0f);
        const float var  = r1[0] * (1.0f / 65536.0f) - mean * mean;
        stats[2 * seg]     = mean;
        stats[2 * seg + 1] = rsqrtf(var + 1e-5f);
    }
}

// ---------------------------------------------------------------------------
// Kernel 2: apply GroupNorm affine, fp32 -> bf16, 4 elems / thread.
// ---------------------------------------------------------------------------
__global__ __launch_bounds__(256) void gn_apply_kernel(
    const float* __restrict__ x, const float* __restrict__ stats,
    const float* __restrict__ gamma, const float* __restrict__ beta,
    bf16* __restrict__ xn) {
    const size_t i = ((size_t)blockIdx.x * 256 + threadIdx.x) * 4;
    const int c  = (int)((i >> 11) & (NC - 1));
    const int bg = (int)(i >> 16);
    const float mean = stats[2 * bg];
    const float rstd = stats[2 * bg + 1];
    const float g = gamma[c] * rstd;
    const float o = beta[c] - mean * g;
    const float4 v = *(const float4*)(x + i);
    union { bf16 h[4]; uint2 u; } pk;
    pk.h[0] = (bf16)(v.x * g + o);
    pk.h[1] = (bf16)(v.y * g + o);
    pk.h[2] = (bf16)(v.z * g + o);
    pk.h[3] = (bf16)(v.w * g + o);
    *(uint2*)(xn + i) = pk.u;
}

// ---------------------------------------------------------------------------
// GEMM config: tile 128x256xBK64, 8 waves (2M x 4N), wave tile 64x64.
// ---------------------------------------------------------------------------
#define BM 128
#define BN 256
#define BK 64
#define AST 72   // padded stride: 144B rows, 16B aligned, odd dword phase
#define BST 72

// ---------------------------------------------------------------------------
// Kernel 3: QKV GEMM (software pipelined).  Y = W*Xn + bias, bf16 out.
// ---------------------------------------------------------------------------
__global__ __launch_bounds__(256) void qkv_gemm_kernel(
    const float* __restrict__ W, const float* __restrict__ bias,
    const bf16* __restrict__ X, bf16* __restrict__ Y) {
    const int b   = blockIdx.z;
    const int m0  = blockIdx.y * BM;
    const int n0  = blockIdx.x * BN;
    const int tid = threadIdx.x;
    const int wave = tid >> 5, lane = tid & 31;
    const int wm = wave & 1, wn = wave >> 1;

    __shared__ bf16 As[BM][AST];
    __shared__ bf16 Bt[BN][BST];

    const bf16* Xb = X + (size_t)b * NC * NT;
    const int arow = tid >> 1;
    const int acol = (tid & 1) * 32;

    float4   areg[8];
    unsigned breg[32];

    // preload k-tile 0 into registers
    {
        const float* src = W + (size_t)(m0 + arow) * NC + acol;
#pragma unroll
        for (int j = 0; j < 8; ++j) areg[j] = *(const float4*)(src + 4 * j);
#pragma unroll
        for (int p = 0; p < 32; ++p) {
            const int d = tid + p * 256;
            const int k = d >> 7, nu = d & 127;
            breg[p] = *((const unsigned*)(Xb + (size_t)k * NT + n0) + nu);
        }
    }

    v8f acc[4][4] = {};

    for (int kk = 0; kk < NC; kk += BK) {
        // commit staged registers to LDS
#pragma unroll
        for (int j = 0; j < 8; ++j) {
            As[arow][acol + 4 * j + 0] = (bf16)areg[j].x;
            As[arow][acol + 4 * j + 1] = (bf16)areg[j].y;
            As[arow][acol + 4 * j + 2] = (bf16)areg[j].z;
            As[arow][acol + 4 * j + 3] = (bf16)areg[j].w;
        }
#pragma unroll
        for (int p = 0; p < 32; ++p) {
            const int d = tid + p * 256;
            const int k = d >> 7, nu = d & 127;
            union { unsigned u; bf16 h[2]; } w2;
            w2.u = breg[p];
            Bt[2 * nu + 0][k] = w2.h[0];
            Bt[2 * nu + 1][k] = w2.h[1];
        }
        __syncthreads();

        // issue next tile's global loads (overlap with WMMA below)
        if (kk + BK < NC) {
            const float* src = W + (size_t)(m0 + arow) * NC + (kk + BK) + acol;
#pragma unroll
            for (int j = 0; j < 8; ++j) areg[j] = *(const float4*)(src + 4 * j);
#pragma unroll
            for (int p = 0; p < 32; ++p) {
                const int d = tid + p * 256;
                const int k = d >> 7, nu = d & 127;
                breg[p] =
                    *((const unsigned*)(Xb + (size_t)(kk + BK + k) * NT + n0) + nu);
            }
        }

#pragma unroll
        for (int ks = 0; ks < BK; ks += 32) {
            v16bf bfr[4];
#pragma unroll
            for (int j = 0; j < 4; ++j)
                bfr[j] = load_b_frag_t<BST>(&Bt[0][0], ks, wn * 64 + j * 16);
#pragma unroll
            for (int i = 0; i < 4; ++i) {
                v16bf afr = load_a_frag<AST>(&As[0][0], wm * 64 + i * 16, ks);
#pragma unroll
                for (int j = 0; j < 4; ++j)
                    acc[i][j] = wmma_bf16(afr, bfr[j], acc[i][j]);
            }
        }
        __syncthreads();
    }

    const int moff = (lane < 16) ? 0 : 8;
#pragma unroll
    for (int i = 0; i < 4; ++i) {
#pragma unroll
        for (int j = 0; j < 4; ++j) {
            const int n = n0 + wn * 64 + j * 16 + (lane & 15);
#pragma unroll
            for (int r = 0; r < 8; ++r) {
                const int m = m0 + wm * 64 + i * 16 + r + moff;
                Y[(size_t)b * 3 * NC * NT + (size_t)m * NT + n] =
                    (bf16)(acc[i][j][r] + bias[m]);
            }
        }
    }
}

// ---------------------------------------------------------------------------
// Kernel 4: flash attention, software-pipelined K/V tiles.
// Grid (T/128, B*H); 8 waves, 16 t-rows each; s blocks of 64.
// ---------------------------------------------------------------------------
__global__ __launch_bounds__(256) void attn_kernel(
    const bf16* __restrict__ qkv, bf16* __restrict__ out) {
    const int t0 = blockIdx.x * 128;
    const int bh = blockIdx.y;
    const int b = bh >> 4, h = bh & 15;
    const int tid = threadIdx.x, wave = tid >> 5, lane = tid & 31;

    const bf16* q = qkv + ((size_t)b * 3 * NC + h * CH) * NT;
    const bf16* k = qkv + ((size_t)b * 3 * NC + NC + h * CH) * NT;
    const bf16* v = qkv + ((size_t)b * 3 * NC + 2 * NC + h * CH) * NT;

    __shared__ bf16 Qs[128][72];     // A tile for S (row-major [t][c])
    __shared__ bf16 Kt[64][72];      // B tile for S (N-major [s][c])
    __shared__ bf16 Vt[64][72];      // B tile for O (N-major [c][s], direct)
    __shared__ bf16 Ps[8][16][72];   // per-wave P staging (A tile for O)

    const int sc = tid >> 5;         // not used; keep wave var below
    (void)sc;

    // Q transposed: 4096 dwords, 16 / thread
#pragma unroll
    for (int p = 0; p < 16; ++p) {
        const int d  = tid + p * 256;
        const int c  = d >> 6;
        const int tu = d & 63;
        union { unsigned u; bf16 h[2]; } w;
        w.u = *((const unsigned*)(q + (size_t)c * NT + t0) + tu);
        Qs[2 * tu + 0][c] = w.h[0];
        Qs[2 * tu + 1][c] = w.h[1];
    }

    const int krow = tid >> 5;       // staging coords for K/V tiles
    // each thread handles 8 dwords of K and 8 of V per s-block
    unsigned kreg[8], vreg[8];
    {
#pragma unroll
        for (int p = 0; p < 8; ++p) {
            const int d  = tid + p * 256;
            const int c  = d >> 5;
            const int su = d & 31;
            kreg[p] = *((const unsigned*)(k + (size_t)c * NT) + su);
            vreg[p] = *((const unsigned*)(v + (size_t)c * NT) + su);
        }
    }
    (void)krow;

    float rmax[8], rsum[8];
#pragma unroll
    for (int r = 0; r < 8; ++r) { rmax[r] = -3.0e38f; rsum[r] = 0.f; }
    v8f oacc[4] = {};
    __syncthreads();

    for (int s0 = 0; s0 < NT; s0 += 64) {
        // commit staged K (transposed) and V (direct) to LDS
#pragma unroll
        for (int p = 0; p < 8; ++p) {
            const int d  = tid + p * 256;
            const int c  = d >> 5;
            const int su = d & 31;
            union { unsigned u; bf16 h[2]; } w;
            w.u = kreg[p];
            Kt[2 * su + 0][c] = w.h[0];
            Kt[2 * su + 1][c] = w.h[1];
            *((unsigned*)(&Vt[c][0]) + su) = vreg[p];
        }
        __syncthreads();

        // issue next s-block's global loads (overlap with compute)
        if (s0 + 64 < NT) {
#pragma unroll
            for (int p = 0; p < 8; ++p) {
                const int d  = tid + p * 256;
                const int c  = d >> 5;
                const int su = d & 31;
                kreg[p] = *((const unsigned*)(k + (size_t)c * NT + s0 + 64) + su);
                vreg[p] = *((const unsigned*)(v + (size_t)c * NT + s0 + 64) + su);
            }
        }

        // S = Q^T K
        v8f sacc[4] = {};
#pragma unroll
        for (int ks = 0; ks < CH; ks += 32) {
            v16bf aq = load_a_frag<72>(&Qs[0][0], wave * 16, ks);
#pragma unroll
            for (int j = 0; j < 4; ++j) {
                v16bf bk = load_b_frag_t<72>(&Kt[0][0], ks, j * 16);
                sacc[j] = wmma_bf16(aq, bk, sacc[j]);
            }
        }

        // online softmax (16-lane reductions within wave halves)
#pragma unroll
        for (int r = 0; r < 8; ++r) {
            float bm = -3.0e38f;
#pragma unroll
            for (int j = 0; j < 4; ++j) bm = fmaxf(bm, sacc[j][r] * 0.125f);
#pragma unroll
            for (int msk = 1; msk < 16; msk <<= 1)
                bm = fmaxf(bm, __shfl_xor(bm, msk, 32));
            const float nm = fmaxf(rmax[r], bm);
            const float cf = __expf(rmax[r] - nm);
            float psum = 0.f;
#pragma unroll
            for (int j = 0; j < 4; ++j) {
                const float p = __expf(sacc[j][r] * 0.125f - nm);
                sacc[j][r] = p;
                psum += p;
            }
#pragma unroll
            for (int msk = 1; msk < 16; msk <<= 1)
                psum += __shfl_xor(psum, msk, 32);
            rsum[r] = rsum[r] * cf + psum;
            rmax[r] = nm;
#pragma unroll
            for (int j = 0; j < 4; ++j) oacc[j][r] *= cf;
        }

        // stage P into this wave's private LDS region
        const int mrow = (lane < 16) ? 0 : 8;
#pragma unroll
        for (int j = 0; j < 4; ++j) {
            const int col = j * 16 + (lane & 15);
#pragma unroll
            for (int r = 0; r < 8; ++r)
                Ps[wave][mrow + r][col] = (bf16)sacc[j][r];
        }
        __builtin_amdgcn_wave_barrier();
        asm volatile("s_wait_dscnt 0" ::: "memory");  // wave-local LDS W->R

        // O += P * V^T
#pragma unroll
        for (int ks = 0; ks < 64; ks += 32) {
            v16bf ap = load_a_frag<72>(&Ps[wave][0][0], 0, ks);
#pragma unroll
            for (int j = 0; j < 4; ++j) {
                v16bf bv = load_b_frag_t<72>(&Vt[0][0], ks, j * 16);
                oacc[j] = wmma_bf16(ap, bv, oacc[j]);
            }
        }
        __syncthreads();
    }

    const int moff = (lane < 16) ? 0 : 8;
#pragma unroll
    for (int r = 0; r < 8; ++r) {
        const float inv = 1.0f / rsum[r];
        const int t = t0 + wave * 16 + r + moff;
#pragma unroll
        for (int j = 0; j < 4; ++j) {
            const int c = j * 16 + (lane & 15);
            out[((size_t)b * NC + h * CH + c) * NT + t] = (bf16)(oacc[j][r] * inv);
        }
    }
}

// ---------------------------------------------------------------------------
// Kernel 5: projection GEMM + bias + residual, fp32 out (pipelined).
// ---------------------------------------------------------------------------
__global__ __launch_bounds__(256) void proj_gemm_kernel(
    const float* __restrict__ W, const float* __restrict__ bias,
    const bf16* __restrict__ X, const float* __restrict__ xres,
    float* __restrict__ out) {
    const int b   = blockIdx.z;
    const int m0  = blockIdx.y * BM;
    const int n0  = blockIdx.x * BN;
    const int tid = threadIdx.x;
    const int wave = tid >> 5, lane = tid & 31;
    const int wm = wave & 1, wn = wave >> 1;

    __shared__ bf16 As[BM][AST];
    __shared__ bf16 Bt[BN][BST];

    const bf16* Xb = X + (size_t)b * NC * NT;
    const int arow = tid >> 1;
    const int acol = (tid & 1) * 32;

    float4   areg[8];
    unsigned breg[32];
    {
        const float* src = W + (size_t)(m0 + arow) * NC + acol;
#pragma unroll
        for (int j = 0; j < 8; ++j) areg[j] = *(const float4*)(src + 4 * j);
#pragma unroll
        for (int p = 0; p < 32; ++p) {
            const int d = tid + p * 256;
            const int k = d >> 7, nu = d & 127;
            breg[p] = *((const unsigned*)(Xb + (size_t)k * NT + n0) + nu);
        }
    }

    v8f acc[4][4] = {};

    for (int kk = 0; kk < NC; kk += BK) {
#pragma unroll
        for (int j = 0; j < 8; ++j) {
            As[arow][acol + 4 * j + 0] = (bf16)areg[j].x;
            As[arow][acol + 4 * j + 1] = (bf16)areg[j].y;
            As[arow][acol + 4 * j + 2] = (bf16)areg[j].z;
            As[arow][acol + 4 * j + 3] = (bf16)areg[j].w;
        }
#pragma unroll
        for (int p = 0; p < 32; ++p) {
            const int d = tid + p * 256;
            const int k = d >> 7, nu = d & 127;
            union { unsigned u; bf16 h[2]; } w2;
            w2.u = breg[p];
            Bt[2 * nu + 0][k] = w2.h[0];
            Bt[2 * nu + 1][k] = w2.h[1];
        }
        __syncthreads();

        if (kk + BK < NC) {
            const float* src = W + (size_t)(m0 + arow) * NC + (kk + BK) + acol;
#pragma unroll
            for (int j = 0; j < 8; ++j) areg[j] = *(const float4*)(src + 4 * j);
#pragma unroll
            for (int p = 0; p < 32; ++p) {
                const int d = tid + p * 256;
                const int k = d >> 7, nu = d & 127;
                breg[p] =
                    *((const unsigned*)(Xb + (size_t)(kk + BK + k) * NT + n0) + nu);
            }
        }

#pragma unroll
        for (int ks = 0; ks < BK; ks += 32) {
            v16bf bfr[4];
#pragma unroll
            for (int j = 0; j < 4; ++j)
                bfr[j] = load_b_frag_t<BST>(&Bt[0][0], ks, wn * 64 + j * 16);
#pragma unroll
            for (int i = 0; i < 4; ++i) {
                v16bf afr = load_a_frag<AST>(&As[0][0], wm * 64 + i * 16, ks);
#pragma unroll
                for (int j = 0; j < 4; ++j)
                    acc[i][j] = wmma_bf16(afr, bfr[j], acc[i][j]);
            }
        }
        __syncthreads();
    }

    const int moff = (lane < 16) ? 0 : 8;
#pragma unroll
    for (int i = 0; i < 4; ++i) {
#pragma unroll
        for (int j = 0; j < 4; ++j) {
            const int n = n0 + wn * 64 + j * 16 + (lane & 15);
#pragma unroll
            for (int r = 0; r < 8; ++r) {
                const int m = m0 + wm * 64 + i * 16 + r + moff;
                const size_t idx = (size_t)b * NC * NT + (size_t)m * NT + n;
                out[idx] = xres[idx] + bias[m] + acc[i][j][r];
            }
        }
    }
}

// ---------------------------------------------------------------------------
extern "C" void kernel_launch(void* const* d_in, const int* in_sizes, int n_in,
                              void* d_out, int out_size, void* d_ws,
                              size_t ws_size, hipStream_t stream) {
    (void)in_sizes; (void)n_in; (void)out_size; (void)ws_size;
    const float* x      = (const float*)d_in[0];
    const float* gns    = (const float*)d_in[1];
    const float* gnb    = (const float*)d_in[2];
    const float* qkv_w  = (const float*)d_in[3];
    const float* qkv_b  = (const float*)d_in[4];
    const float* proj_w = (const float*)d_in[5];
    const float* proj_b = (const float*)d_in[6];
    float* out = (float*)d_out;

    char* ws = (char*)d_ws;
    const size_t xn_bytes  = (size_t)NB * NC * NT * 2;       // 16 MB
    const size_t qkv_bytes = 3 * xn_bytes;                   // 48 MB
    float* stats = (float*)ws;
    bf16*  xn    = (bf16*)(ws + 4096);
    bf16*  qkv   = (bf16*)(ws + 4096 + xn_bytes);
    bf16*  attn  = (bf16*)(ws + 4096 + xn_bytes + qkv_bytes);

    gn_stats_kernel<<<NB * NG, 256, 0, stream>>>(x, stats);
    gn_apply_kernel<<<(NB * NC * NT) / (256 * 4), 256, 0, stream>>>(
        x, stats, gns, gnb, xn);
    qkv_gemm_kernel<<<dim3(NT / BN, (3 * NC) / BM, NB), 256, 0, stream>>>(
        qkv_w, qkv_b, xn, qkv);
    attn_kernel<<<dim3(NT / 128, NB * NH), 256, 0, stream>>>(qkv, attn);
    proj_gemm_kernel<<<dim3(NT / BN, NC / BM, NB), 256, 0, stream>>>(
        proj_w, proj_b, attn, x, out);
}
